// RRPRAMAttn_50440095924314
// MI455X (gfx1250) — matrix operand order repaired and checked by reading, not compiled, not run on hardware
//
#include <hip/hip_runtime.h>
#include <stdint.h>

// Problem constants (from reference): B,T,E,H,D = 8,2048,1024,16,64
#define B_  8
#define T_  2048
#define E_  1024
#define H_  16
#define D_  64
#define M_  (B_ * T_)    // 16384 rows for both GEMMs
#define HD_ (H_ * D_)    // 1024

typedef __attribute__((ext_vector_type(16))) __bf16 v16bf;
typedef __attribute__((ext_vector_type(8)))  float  v8f;

union FragU {
    v16bf v;
    uint4 q[2];
};

__device__ __forceinline__ unsigned short f32_bf16_rne(float f) {
    union { float f; unsigned int u; } x;
    x.f = f;
    unsigned int u = x.u;
    u += 0x7FFFu + ((u >> 16) & 1u);   // round-to-nearest-even
    return (unsigned short)(u >> 16);
}

// ---------------------------------------------------------------------------
// CDNA5 async copy global->LDS (ASYNCcnt-tracked). lds_off is the low 32 bits
// of a generic LDS pointer (== LDS byte address per the aperture mapping).
// ---------------------------------------------------------------------------
__device__ __forceinline__ void async_g2l_b128(unsigned lds_off, const void* gaddr) {
    asm volatile("global_load_async_to_lds_b128 %0, %1, off"
                 :: "v"(lds_off), "v"((unsigned long long)gaddr)
                 : "memory");
}
#define S_WAIT_ASYNCCNT(n) asm volatile("s_wait_asynccnt " #n ::: "memory")

// ---------------------------------------------------------------------------
// Elementwise f32 -> bf16 (n must be a multiple of 4)
// ---------------------------------------------------------------------------
__global__ void __launch_bounds__(256)
cvt_f32_bf16_kernel(const float* __restrict__ in, unsigned short* __restrict__ out,
                    int n4) {
    int i = blockIdx.x * 256 + threadIdx.x;
    if (i >= n4) return;
    float4 f = ((const float4*)in)[i];
    ushort4 o;
    o.x = f32_bf16_rne(f.x);
    o.y = f32_bf16_rne(f.y);
    o.z = f32_bf16_rne(f.z);
    o.w = f32_bf16_rne(f.w);
    ((ushort4*)out)[i] = o;
}

// ---------------------------------------------------------------------------
// f32 [1024 x 1024] (n-major) -> bf16 transposed [1024 x 1024] (k-major)
// outT[k*1024 + n] = in[n*1024 + k]
// ---------------------------------------------------------------------------
__global__ void __launch_bounds__(256)
cvt_transpose_f32_bf16_kernel(const float* __restrict__ in,
                              unsigned short* __restrict__ outT) {
    int gid = blockIdx.x * 256 + threadIdx.x;   // 1M elements
    int k = gid & 1023;
    int n = gid >> 10;
    outT[(size_t)k * 1024 + n] = f32_bf16_rne(in[(size_t)n * 1024 + k]);
}

// ---------------------------------------------------------------------------
// LDS tile geometry (bf16 elements), padded to spread LDS banks:
//   A tile: 128 rows x 32 k, row stride 40  (80 B)
//   B tile:  32 k   x 128 n, row stride 136 (272 B)
// ---------------------------------------------------------------------------
#define LDS_A_STRIDE 40
#define LDS_B_STRIDE 136
#define LDS_A_ELEMS  (128 * LDS_A_STRIDE)   // 5120
#define LDS_B_ELEMS  (32 * LDS_B_STRIDE)    // 4352

// A-fragment from LDS tile: ISA 7.12.2 16x32 bf16 A layout.
__device__ __forceinline__ v16bf lds_fragA(const unsigned short* tile, int mrow, int lane) {
    int m  = mrow + (lane & 15);
    int hf = lane >> 4;
    const unsigned short* p = tile + m * LDS_A_STRIDE + hf * 8;
    FragU fr;
    fr.q[0] = *(const uint4*)(p);
    fr.q[1] = *(const uint4*)(p + 16);
    return fr.v;
}

// B-fragment from LDS tile: N = col0 + (lane&15); elements s<8 -> K=8*hf+s,
// s>=8 -> K=16+8*hf+(s-8) (same K order as A).
__device__ __forceinline__ v16bf lds_fragB(const unsigned short* tile, int col0, int lane) {
    int hf = lane >> 4;
    const unsigned short* p = tile + (hf * 8) * LDS_B_STRIDE + col0 + (lane & 15);
    unsigned short e[16];
#pragma unroll
    for (int s = 0; s < 8; ++s) {
        e[s]     = p[s * LDS_B_STRIDE];
        e[8 + s] = p[(16 + s) * LDS_B_STRIDE];
    }
    FragU fr;
    __builtin_memcpy(&fr, e, sizeof(e));
    return fr.v;
}

// Issue one double-buffer stage: 4 async b128 loads per thread
// (A tile 8 KB = 512 chunks, B tile 8 KB = 512 chunks; 256 threads).
__device__ __forceinline__ void stage_async(const unsigned short* __restrict__ A,
                                            const unsigned short* __restrict__ Bt,
                                            unsigned ldsA_off, unsigned ldsB_off,
                                            int Mb, int Nb, int k0, int tid) {
#pragma unroll
    for (int i = 0; i < 2; ++i) {
        int c   = tid + i * 256;        // [0,512)
        int row = c >> 2;               // 4 x 16B chunks per 64 B A row
        int ko  = c & 3;
        const unsigned short* g = A + (((size_t)(Mb + row)) << 10) + k0 + ko * 8;
        async_g2l_b128(ldsA_off + (unsigned)(row * LDS_A_STRIDE + ko * 8) * 2u, g);
    }
#pragma unroll
    for (int i = 0; i < 2; ++i) {
        int c  = tid + i * 256;         // [0,512)
        int kr = c >> 4;                // 16 x 16B chunks per 256 B B row
        int co = c & 15;
        const unsigned short* g = Bt + (((size_t)(k0 + kr)) << 10) + Nb + co * 8;
        async_g2l_b128(ldsB_off + (unsigned)(kr * LDS_B_STRIDE + co * 8) * 2u, g);
    }
}

// ---------------------------------------------------------------------------
// WMMA GEMM: C[M x 1024] = A(bf16)[M x K] * B(bf16)[K x 1024], f32 accum.
// Block = 256 threads = 8 waves; block tile 128x128; wave tile 32x64
// (2 A-frags x 4 B-frags -> 8 v_wmma per 32-wide K step).
// Double-buffered LDS staging via global_load_async_to_lds_b128 / ASYNCcnt.
// MODE 0: scatter into v layout [B,H,T,D]; MODE 1: plain row-major [M x 1024].
// ---------------------------------------------------------------------------
template <int MODE>
__global__ void __launch_bounds__(256)
wmma_gemm_bf16_kernel(const unsigned short* __restrict__ A,
                      const unsigned short* __restrict__ Bt,
                      float* __restrict__ out, int K) {
    __shared__ unsigned short ldsA[2][LDS_A_ELEMS];
    __shared__ unsigned short ldsB[2][LDS_B_ELEMS];

    const int tid   = threadIdx.x;
    const int lane  = tid & 31;
    const int wave  = tid >> 5;
    const int waveM = wave & 3;            // 4 waves along M
    const int waveN = wave >> 2;           // 2 waves along N
    const int Mb = blockIdx.y * 128;
    const int Nb = blockIdx.x * 128;

    const unsigned aOff0 = (unsigned)(size_t)(void*)&ldsA[0][0];
    const unsigned aOff1 = (unsigned)(size_t)(void*)&ldsA[1][0];
    const unsigned bOff0 = (unsigned)(size_t)(void*)&ldsB[0][0];
    const unsigned bOff1 = (unsigned)(size_t)(void*)&ldsB[1][0];

    const v8f zero = {0.f, 0.f, 0.f, 0.f, 0.f, 0.f, 0.f, 0.f};
    v8f acc[2][4];
#pragma unroll
    for (int mi = 0; mi < 2; ++mi)
#pragma unroll
        for (int ni = 0; ni < 4; ++ni) acc[mi][ni] = zero;

    // prologue: stage k0 = 0 into buffer 0
    stage_async(A, Bt, aOff0, bOff0, Mb, Nb, 0, tid);

    for (int k0 = 0; k0 < K; k0 += 32) {
        const int  cur  = (k0 >> 5) & 1;
        const bool more = (k0 + 32) < K;

        // issue next stage into the other buffer (its last readers finished at
        // the trailing barrier of the previous iteration)
        if (more) {
            stage_async(A, Bt, cur ? aOff0 : aOff1, cur ? bOff0 : bOff1,
                        Mb, Nb, k0 + 32, tid);
            S_WAIT_ASYNCCNT(0x4);   // retire current stage (async loads in order)
        } else {
            S_WAIT_ASYNCCNT(0x0);
        }
        __syncthreads();            // current tiles visible to all waves

        const unsigned short* tA = &ldsA[cur][0];
        const unsigned short* tB = &ldsB[cur][0];
        v16bf af[2], bfr[4];
#pragma unroll
        for (int mi = 0; mi < 2; ++mi)
            af[mi] = lds_fragA(tA, waveM * 32 + mi * 16, lane);
#pragma unroll
        for (int ni = 0; ni < 4; ++ni)
            bfr[ni] = lds_fragB(tB, waveN * 64 + ni * 16, lane);
#pragma unroll
        for (int mi = 0; mi < 2; ++mi)
#pragma unroll
            for (int ni = 0; ni < 4; ++ni)
                acc[mi][ni] = __builtin_amdgcn_wmma_f32_16x16x32_bf16(
                    false, af[mi], false, bfr[ni], (short)0, acc[mi][ni],
                    false, false);

        __syncthreads();            // all reads done before buffer is rewritten
    }

    // C/D layout: VGPR r, lanes 0-15 -> M=r, lanes 16-31 -> M=r+8; N = lane&15.
    const int hf = lane >> 4;
    const int nl = lane & 15;
#pragma unroll
    for (int mi = 0; mi < 2; ++mi) {
#pragma unroll
        for (int ni = 0; ni < 4; ++ni) {
#pragma unroll
            for (int r = 0; r < 8; ++r) {
                int row = Mb + waveM * 32 + mi * 16 + hf * 8 + r;   // = b*T + t
                int col = Nb + waveN * 64 + ni * 16 + nl;
                float val = acc[mi][ni][r];
                if (MODE == 0) {
                    int b = row >> 11, t = row & (T_ - 1);
                    int h = col >> 6,  d = col & (D_ - 1);
                    out[(((size_t)(b * H_ + h)) * T_ + t) * D_ + d] = val;
                } else {
                    out[(size_t)row * 1024 + col] = val;
                }
            }
        }
    }
}

// ---------------------------------------------------------------------------
// a[b,h,t] = (1/sqrt(D)) * sum_e x[b,t,e] * wr[h,e,t]
// One thread per (b,h,t); x read contiguous per thread (float4), wr reads
// lane-coalesced (consecutive t across lanes).
// ---------------------------------------------------------------------------
__global__ void __launch_bounds__(256)
logits_kernel(const float* __restrict__ x, const float* __restrict__ wr,
              float* __restrict__ a) {
    int gid = blockIdx.x * 256 + threadIdx.x;     // [0, B*H*T)
    int t  = gid & (T_ - 1);
    int bh = gid >> 11;
    int b = bh >> 4;
    int h = bh & (H_ - 1);
    const float* xrow = x  + ((size_t)b * T_ + t) * E_;
    const float* wcol = wr + (size_t)h * E_ * T_ + t;
    float acc = 0.f;
#pragma unroll 4
    for (int e = 0; e < E_; e += 4) {
        float4 xv = *(const float4*)(xrow + e);
        acc += xv.x * wcol[(size_t)(e + 0) * T_];
        acc += xv.y * wcol[(size_t)(e + 1) * T_];
        acc += xv.z * wcol[(size_t)(e + 2) * T_];
        acc += xv.w * wcol[(size_t)(e + 3) * T_];
    }
    a[gid] = acc * 0.125f;   // 1/sqrt(64)
}

// ---------------------------------------------------------------------------
// Per (b,h): amax over T, then inclusive prefix scan of w and w*v over T,
// out[t,d] = num/den, stored bf16 into attn[B,T,H,D] for GEMM2.
// 256 threads = 8 wave32 chunks of 256 t each; two-pass chunked scan.
// Each lane owns d = lane and d = lane+32.
// ---------------------------------------------------------------------------
__global__ void __launch_bounds__(256)
scan_kernel(const float* __restrict__ a, const float* __restrict__ v,
            unsigned short* __restrict__ attn) {
    __shared__ float red[256];
    __shared__ float sden[8];
    __shared__ float snum[8 * 64];

    const int bh   = blockIdx.x;
    const int b    = bh >> 4, h = bh & (H_ - 1);
    const int tid  = threadIdx.x;
    const int wave = tid >> 5, lane = tid & 31;
    const float* arow = a + (size_t)bh * T_;
    const float* vrow = v + (size_t)bh * T_ * D_;

    // global max over the (b,h) logit row
    float m = -3.402823466e38f;
    for (int t = tid; t < T_; t += 256) m = fmaxf(m, arow[t]);
    red[tid] = m;
    __syncthreads();
    for (int s = 128; s > 0; s >>= 1) {
        if (tid < s) red[tid] = fmaxf(red[tid], red[tid + s]);
        __syncthreads();
    }
    const float amax = red[0];
    __syncthreads();

    // pass 1: per-chunk totals
    const int t0 = wave * 256;
    float den = 0.f, n0 = 0.f, n1 = 0.f;
    for (int t = t0; t < t0 + 256; ++t) {
        float w = __expf(arow[t] - amax);
        const float* vp = vrow + (size_t)t * D_;
        den += w;
        n0 += w * vp[lane];
        n1 += w * vp[lane + 32];
    }
    if (lane == 0) sden[wave] = den;
    snum[wave * 64 + lane]      = n0;
    snum[wave * 64 + lane + 32] = n1;
    __syncthreads();

    // exclusive prefix over the 8 chunks
    float offd = 0.f, o0 = 0.f, o1 = 0.f;
    for (int j = 0; j < wave; ++j) {
        offd += sden[j];
        o0   += snum[j * 64 + lane];
        o1   += snum[j * 64 + lane + 32];
    }

    // pass 2: running inclusive scan inside the chunk, normalized write
    float accd = offd, a0 = o0, a1 = o1;
    for (int t = t0; t < t0 + 256; ++t) {
        float w = __expf(arow[t] - amax);
        const float* vp = vrow + (size_t)t * D_;
        accd += w;
        a0 += w * vp[lane];
        a1 += w * vp[lane + 32];
        float r = 1.0f / accd;
        size_t o = (((size_t)b * T_ + t) * H_ + h) * D_;
        attn[o + lane]      = f32_bf16_rne(a0 * r);
        attn[o + lane + 32] = f32_bf16_rne(a1 * r);
    }
}

// ---------------------------------------------------------------------------
// Launch: cvt(x), cvtT(wv), cvtT(wo) -> GEMM1 -> logits -> scan -> GEMM2
// Workspace layout (133 MiB total):
//   [0,64MiB)    v      f32 [B,H,T,D]
//   [64,65MiB)   a      f32 [B,H,T]
//   [65,97MiB)   x_bf16        [M x E]
//   [97,99MiB)   wvT_bf16      [E x HD]
//   [99,101MiB)  woT_bf16      [HD x E]
//   [101,133MiB) attn_bf16     [M x HD]
// ---------------------------------------------------------------------------
extern "C" void kernel_launch(void* const* d_in, const int* in_sizes, int n_in,
                              void* d_out, int out_size, void* d_ws, size_t ws_size,
                              hipStream_t stream) {
    const float* x  = (const float*)d_in[0];
    const float* wv = (const float*)d_in[1];
    const float* wr = (const float*)d_in[2];
    const float* wo = (const float*)d_in[3];
    float* out = (float*)d_out;

    char* ws = (char*)d_ws;
    const size_t MiB = 1024 * 1024;
    float*          v_f32   = (float*)(ws);
    float*          a_f32   = (float*)(ws + 64 * MiB);
    unsigned short* x_bf    = (unsigned short*)(ws + 65 * MiB);
    unsigned short* wvT_bf  = (unsigned short*)(ws + 97 * MiB);
    unsigned short* woT_bf  = (unsigned short*)(ws + 99 * MiB);
    unsigned short* attn_bf = (unsigned short*)(ws + 101 * MiB);

    // precision conversions (one pass each; removes cvt VALU from GEMM loops)
    {
        int n4 = (B_ * T_ * E_) / 4;
        cvt_f32_bf16_kernel<<<(n4 + 255) / 256, 256, 0, stream>>>(x, x_bf, n4);
        cvt_transpose_f32_bf16_kernel<<<(1024 * 1024) / 256, 256, 0, stream>>>(wv, wvT_bf);
        cvt_transpose_f32_bf16_kernel<<<(1024 * 1024) / 256, 256, 0, stream>>>(wo, woT_bf);
    }

    // v = x @ wv^T  -> [B,H,T,D] f32
    {
        dim3 grid(HD_ / 128, M_ / 128);
        wmma_gemm_bf16_kernel<0><<<grid, 256, 0, stream>>>(x_bf, wvT_bf, v_f32, E_);
    }

    // a[b,h,t] (f32, full precision dot)
    logits_kernel<<<(B_ * H_ * T_) / 256, 256, 0, stream>>>(x, wr, a_f32);

    // causal-softmax prefix scan -> attn bf16 [B,T,H,D]
    scan_kernel<<<B_ * H_, 256, 0, stream>>>(a_f32, v_f32, attn_bf);

    // out = attn @ wo^T -> [B,T,E] f32
    {
        dim3 grid(E_ / 128, M_ / 128);
        wmma_gemm_bf16_kernel<1><<<grid, 256, 0, stream>>>(attn_bf, woT_bf, out, HD_);
    }
}
